// GATLayer_89936615178507
// MI455X (gfx1250) — matrix operand order repaired and checked by reading, not compiled, not run on hardware
//
#include <hip/hip_runtime.h>
#include <hip/hip_bf16.h>

// ---------------------------------------------------------------------------
// GAT layer for MI455X (gfx1250, wave32, WMMA).
// B=4, N=2048, C_IN=256, H=8, C_HEAD=32, concat -> C_OUT=256.
//
// d_in: [0]=x f32[4,2048,256]  [1]=G i32[4,2048,2048]
//       [2]=proj_kernel f32[256,256]  [3]=proj_bias f32[256]
//       [4]=attn_fst f32[8,32] [5]=attn_snd f32[8,32]
// d_out: f32[4,2048,256]
//
// ws layout (bytes): xb bf16 @0 (4MB) | wb bf16 @4194304 (128KB)
//   hb bf16 @4325376 (4MB) | lf_t f32 @8519680 (256KB)
//   ls_t f32 @8781824 (256KB) | lsmax f32 @9043968 (128B)   total ~9.05MB
// ---------------------------------------------------------------------------

#define B_    4
#define N_    2048
#define CIN_  256
#define H_    8
#define CH_   32
#define ALPHA_ 0.2f

typedef __attribute__((ext_vector_type(16))) __bf16 v16bf;
typedef __attribute__((ext_vector_type(8)))  __bf16 v8bf;
typedef __attribute__((ext_vector_type(8)))  float  v8f;
typedef __attribute__((ext_vector_type(2)))  int    v2i;

// CDNA5 async global->LDS path (ASYNCcnt), if this toolchain exposes it.
#if defined(__has_builtin)
#if __has_builtin(__builtin_amdgcn_global_load_async_to_lds_b64) && \
    __has_builtin(__builtin_amdgcn_s_wait_asynccnt)
#define HAVE_ASYNC_LDS 1
#endif
#endif

__device__ __forceinline__ __bf16 f2bf(float f) {
  unsigned u = __float_as_uint(f);
  u += 0x7fffu + ((u >> 16) & 1u);              // round-to-nearest-even
  unsigned short s = (unsigned short)(u >> 16);
  return __builtin_bit_cast(__bf16, s);
}
__device__ __forceinline__ float bf2f(__bf16 h) {
  unsigned short s = __builtin_bit_cast(unsigned short, h);
  return __uint_as_float(((unsigned)s) << 16);
}

// ---------------- f32 -> bf16 conversion -----------------------------------
__global__ void cvt_f32_bf16(const float* __restrict__ s,
                             __bf16* __restrict__ d, int n) {
  int i = (blockIdx.x * blockDim.x + threadIdx.x) * 4;
  if (i + 3 < n) {
    float4 v = *(const float4*)(s + i);
    d[i + 0] = f2bf(v.x); d[i + 1] = f2bf(v.y);
    d[i + 2] = f2bf(v.z); d[i + 3] = f2bf(v.w);
  } else {
    for (int k = i; k < n; ++k) d[k] = f2bf(s[k]);
  }
}

// ---------------- projection GEMM: h = x@W + b -----------------------------
// One wave computes a 16x16 C tile; K loop over 256 in steps of 32.
// A 16x32 bf16 per-lane layout (ISA 7.12.2): lane<16 -> M=lane, K={0..7,16..23}
// lane>=16 -> M=lane-16, K={8..15,24..31}. B: N = lane&15, same K as A.
__global__ void proj_gemm(const __bf16* __restrict__ xb,
                          const __bf16* __restrict__ wb,
                          const float*  __restrict__ bias,
                          __bf16* __restrict__ hb) {
  int tile = blockIdx.x * 8 + (threadIdx.x >> 5);   // 8192 wave-tiles
  int lane = threadIdx.x & 31;
  int mt = tile >> 4, nt = tile & 15;               // 512 m-tiles x 16 n-tiles
  int m0 = mt * 16, n0 = nt * 16;
  int lm = lane & 15, sel = lane >> 4;

  v8f acc = {};
  const __bf16* arow = xb + (size_t)(m0 + lm) * CIN_;
#pragma unroll
  for (int kk = 0; kk < CIN_; kk += 32) {
    v8bf alo = *(const v8bf*)(arow + kk + sel * 8);
    v8bf ahi = *(const v8bf*)(arow + kk + 16 + sel * 8);
    v16bf a;
#pragma unroll
    for (int e = 0; e < 8; ++e) { a[e] = alo[e]; a[e + 8] = ahi[e]; }
    v16bf bv;
#pragma unroll
    for (int e = 0; e < 8; ++e) {
      bv[e]     = wb[(size_t)(kk + sel * 8 + e) * 256 + n0 + lm];
      bv[e + 8] = wb[(size_t)(kk + 16 + sel * 8 + e) * 256 + n0 + lm];
    }
    acc = __builtin_amdgcn_wmma_f32_16x16x32_bf16(
        false, a, false, bv, (short)0, acc, false, false);
  }
  float bb = bias[n0 + lm];                          // C col = lane&15
#pragma unroll
  for (int r = 0; r < 8; ++r) {
    int row = m0 + (sel ? 8 + r : r);                // C row layout
    hb[(size_t)row * 256 + n0 + lm] = f2bf(acc[r] + bb);
  }
}

// ---------------- per-node attention scores (head-major) -------------------
__global__ void lfls_kernel(const __bf16* __restrict__ hb,
                            const float* __restrict__ afst,
                            const float* __restrict__ asnd,
                            float* __restrict__ lf_t,
                            float* __restrict__ ls_t) {
  int gid = blockIdx.x * blockDim.x + threadIdx.x;   // 65536 threads
  int n = gid & (N_ - 1);
  int t = gid >> 11;
  int hd = t & (H_ - 1);
  int b  = t >> 3;
  const __bf16* p = hb + ((size_t)(b * N_ + n) * 256) + hd * CH_;
  v16bf h0 = *(const v16bf*)(p);
  v16bf h1 = *(const v16bf*)(p + 16);
  float f = 0.f, s = 0.f;
#pragma unroll
  for (int c = 0; c < 16; ++c) {
    float v0 = bf2f(h0[c]);
    float v1 = bf2f(h1[c]);
    f += v0 * afst[hd * CH_ + c] + v1 * afst[hd * CH_ + 16 + c];
    s += v0 * asnd[hd * CH_ + c] + v1 * asnd[hd * CH_ + 16 + c];
  }
  size_t o = (size_t)(b * H_ + hd) * N_ + n;
  lf_t[o] = f;
  ls_t[o] = s;
}

// ---------------- per-(b,head) max of ls (for fixed softmax bound) ---------
__global__ void lsmax_kernel(const float* __restrict__ ls_t,
                             float* __restrict__ lsmax) {
  __shared__ float red[256];
  int t = threadIdx.x;
  const float* p = ls_t + (size_t)blockIdx.x * N_;
  float m = -3.0e38f;
  for (int i = t; i < N_; i += 256) m = fmaxf(m, p[i]);
  red[t] = m;
  __syncthreads();
  for (int s = 128; s > 0; s >>= 1) {
    if (t < s) red[t] = fmaxf(red[t], red[t + s]);
    __syncthreads();
  }
  if (t == 0) lsmax[blockIdx.x] = red[0];
}

// ---------------- fused flash-softmax GAT aggregation ----------------------
// grid = B * N/16 blocks; block = 256 threads = 8 waves; wave w = head w.
// G tile [16 x 32] staged in LDS once per block, shared by all 8 heads.
// Softmax uses the exact fixed bound m_i = lrelu(lf_i + max_j ls_j), so the
// j-loop has no running max / accumulator rescaling.
__global__ void flash_gat(const __bf16* __restrict__ hb,
                          const float* __restrict__ lf_t,
                          const float* __restrict__ ls_t,
                          const float* __restrict__ lsmax,
                          const int*   __restrict__ G,
                          float* __restrict__ out) {
  const int b  = blockIdx.x >> 7;                    // / (N/16)
  const int i0 = (blockIdx.x & 127) * 16;
  const int hd   = threadIdx.x >> 5;
  const int lane = threadIdx.x & 31;
  const int lm = lane & 15, sel = lane >> 4;

  __shared__ __align__(16) int Gt[2][16 * 32];

  // cooperative G tile staging: thread t -> row t/16, two cols (8B = b64)
  const int gr = threadIdx.x >> 4;
  const int gc = (threadIdx.x & 15) * 2;
  const int* gsrc = G + ((size_t)b * N_ + i0 + gr) * N_ + gc;
  int* gdst0 = &Gt[0][gr * 32 + gc];
  int* gdst1 = &Gt[1][gr * 32 + gc];

#if HAVE_ASYNC_LDS
  // builtin signature (from probe diag): (v2i AS1*, v2i AS3*, Ii off, Ii cpol)
#define STAGE_G(dst, src)                                                     \
  __builtin_amdgcn_global_load_async_to_lds_b64(                              \
      (__attribute__((address_space(1))) v2i*)(src),                          \
      (__attribute__((address_space(3))) v2i*)(dst), 0, 0)
#define WAIT_G() __builtin_amdgcn_s_wait_asynccnt(0)
#else
#define STAGE_G(dst, src)                                                     \
  do { int2 v_ = *(const int2*)(src); (dst)[0] = v_.x; (dst)[1] = v_.y; } while (0)
#define WAIT_G() ((void)0)
#endif

  STAGE_G(gdst0, gsrc);                              // chunk 0 -> buffer 0
  WAIT_G();
  __syncthreads();

  const size_t hdbase = (size_t)(b * H_ + hd) * N_;
  const float lf_val = lf_t[hdbase + i0 + lm];       // row score, lane-resident
  // fixed per-row softmax bound: lrelu is monotone, so this is the exact
  // unmasked row max and a safe bound for the masked one.
  const float lg0 = lf_val + lsmax[b * H_ + hd];
  const float m_fix = lg0 > 0.0f ? lg0 : ALPHA_ * lg0;

  float lsum = 0.0f;                                 // this lane's half-row sum
  v8f acc0 = {}, acc1 = {};

  for (int ci = 0; ci < N_ / 32; ++ci) {
    const int j0 = ci * 32;
    const int cur = ci & 1;

    if (j0 + 32 < N_)                                // stage next tile -> LDS
      STAGE_G(cur ? gdst0 : gdst1, gsrc + j0 + 32);
    if (j0 + 64 < N_)                                // L2 prefetch 2 ahead
      __builtin_prefetch(gsrc + j0 + 64, 0, 1);

    // --- this lane's 16 (row lm, K) entries: K = sel*8+{0..7}, 16+sel*8+{0..7}
    const int* gl = &Gt[cur][lm * 32 + sel * 8];
    int4 g0 = *(const int4*)(gl);
    int4 g1 = *(const int4*)(gl + 4);
    int4 g2 = *(const int4*)(gl + 16);
    int4 g3 = *(const int4*)(gl + 20);
    int gv[16] = {g0.x,g0.y,g0.z,g0.w, g1.x,g1.y,g1.z,g1.w,
                  g2.x,g2.y,g2.z,g2.w, g3.x,g3.y,g3.z,g3.w};

    const float* lsp = ls_t + hdbase + j0 + sel * 8;
    float4 s0 = *(const float4*)(lsp);
    float4 s1 = *(const float4*)(lsp + 4);
    float4 s2 = *(const float4*)(lsp + 16);
    float4 s3 = *(const float4*)(lsp + 20);
    float lsv[16] = {s0.x,s0.y,s0.z,s0.w, s1.x,s1.y,s1.z,s1.w,
                     s2.x,s2.y,s2.z,s2.w, s3.x,s3.y,s3.z,s3.w};

    // --- masked leaky-relu -> exp -> bf16 P tile (A-matrix lane ordering)
    v16bf a;
#pragma unroll
    for (int e = 0; e < 16; ++e) {
      float lg = lf_val + lsv[e];
      float lr = lg > 0.0f ? lg : ALPHA_ * lg;
      float p  = (gv[e] != 0) ? __expf(lr - m_fix) : 0.0f;
      lsum += p;
      a[e] = f2bf(p);
    }

    // --- B tiles: col lm (/ 16+lm) of this head, K = sel*8+{0..7}, 16+sel*8+{0..7}
    const __bf16* hrow = hb + ((size_t)(b * N_ + j0) * 256) + hd * CH_ + lm;
    v16bf b0, b1;
#pragma unroll
    for (int e = 0; e < 8; ++e) {
      b0[e]     = hrow[(size_t)(sel * 8 + e) * 256];
      b0[e + 8] = hrow[(size_t)(16 + sel * 8 + e) * 256];
      b1[e]     = hrow[(size_t)(sel * 8 + e) * 256 + 16];
      b1[e + 8] = hrow[(size_t)(16 + sel * 8 + e) * 256 + 16];
    }

    acc0 = __builtin_amdgcn_wmma_f32_16x16x32_bf16(
        false, a, false, b0, (short)0, acc0, false, false);
    acc1 = __builtin_amdgcn_wmma_f32_16x16x32_bf16(
        false, a, false, b1, (short)0, acc1, false, false);

    WAIT_G();                                        // next-tile staging done
    __syncthreads();                                 // protects both buffers
  }

  // --- epilogue: combine half-row sums, divide, write f32 out
  lsum += __shfl_xor(lsum, 16, 32);
#pragma unroll
  for (int r = 0; r < 8; ++r) {
    float llo = __shfl(lsum, r, 32);
    float lhi = __shfl(lsum, r + 8, 32);
    float inv = 1.0f / (sel ? lhi : llo);
    int row = i0 + (sel ? 8 + r : r);
    size_t o = ((size_t)b * N_ + row) * 256 + hd * CH_ + lm;
    out[o]      = acc0[r] * inv;
    out[o + 16] = acc1[r] * inv;
  }
#undef STAGE_G
#undef WAIT_G
}

// ---------------------------------------------------------------------------
extern "C" void kernel_launch(void* const* d_in, const int* in_sizes, int n_in,
                              void* d_out, int out_size, void* d_ws, size_t ws_size,
                              hipStream_t stream) {
  (void)in_sizes; (void)n_in; (void)out_size; (void)ws_size;
  const float* x    = (const float*)d_in[0];
  const int*   G    = (const int*)d_in[1];
  const float* W    = (const float*)d_in[2];
  const float* bias = (const float*)d_in[3];
  const float* afst = (const float*)d_in[4];
  const float* asnd = (const float*)d_in[5];
  float* out = (float*)d_out;

  char* ws = (char*)d_ws;
  __bf16* xb    = (__bf16*)(ws);
  __bf16* wb    = (__bf16*)(ws + 4194304);
  __bf16* hb    = (__bf16*)(ws + 4325376);
  float*  lf_t  = (float*)(ws + 8519680);
  float*  ls_t  = (float*)(ws + 8781824);
  float*  lsmax = (float*)(ws + 9043968);

  const int nx = B_ * N_ * CIN_;     // 2097152
  const int nw = CIN_ * 256;         // 65536

  cvt_f32_bf16<<<(nx / 4 + 255) / 256, 256, 0, stream>>>(x, xb, nx);
  cvt_f32_bf16<<<(nw / 4 + 255) / 256, 256, 0, stream>>>(W, wb, nw);
  proj_gemm<<<1024, 256, 0, stream>>>(xb, wb, bias, hb);       // 8192 tiles / 8 waves
  lfls_kernel<<<(B_ * H_ * N_) / 256, 256, 0, stream>>>(hb, afst, asnd, lf_t, ls_t);
  lsmax_kernel<<<B_ * H_, 256, 0, stream>>>(ls_t, lsmax);
  flash_gat<<<B_ * (N_ / 16), 256, 0, stream>>>(hb, lf_t, ls_t, lsmax, G, out);
}
